// GCN_13709535609170
// MI455X (gfx1250) — matrix operand order repaired and checked by reading, compiled, not run on hardware
//
#include <hip/hip_runtime.h>
#include <hip/hip_bf16.h>
#include <math.h>
#include <stdint.h>

// ---------------- problem constants (from reference) ----------------
#define NN      100000
#define NFEAT   512
#define NHID    128
#define NCLASS  40
#define NCPAD   48
#define NEDGE   3200000

typedef __attribute__((ext_vector_type(16))) __bf16 v16bf;
typedef __attribute__((ext_vector_type(8)))  __bf16 v8bf;
typedef __attribute__((ext_vector_type(8)))  float  v8f;

union BV { v16bf v; v8bf h[2]; };

// async 16-byte global -> LDS copy (per-lane), tracked by ASYNCcnt
__device__ __forceinline__ void async_g2l_b128(uint32_t lds_addr, const void* gaddr) {
  asm volatile("global_load_async_to_lds_b128 %0, %1, off"
               :: "v"(lds_addr), "v"((unsigned long long)gaddr) : "memory");
}
__device__ __forceinline__ void async_wait_all() {
  asm volatile("s_wait_asynccnt 0x0" ::: "memory");
}

// ---------------- prep: transpose weights to bf16 ----------------
// W1t[n*512+k] = W1[k*128+n]   (128 x 512 bf16)
// W2t[n*128+k] = W2[k*40+n] (zero-pad n>=40)  (48 x 128 bf16)
__global__ __launch_bounds__(256) void gcn_prep(const float* __restrict__ W1,
                                                const float* __restrict__ W2,
                                                __bf16* __restrict__ W1t,
                                                __bf16* __restrict__ W2t) {
  int i = blockIdx.x * 256 + threadIdx.x;
  if (i < NHID * NFEAT) {
    int n = i / NFEAT, k = i % NFEAT;
    W1t[i] = (__bf16)W1[k * NHID + n];
  }
  if (i < NCPAD * NHID) {
    int n = i / NHID, k = i % NHID;
    W2t[i] = (__bf16)((n < NCLASS) ? W2[k * NCLASS + n] : 0.0f);
  }
}

// ---------------- init: h = broadcast(b1), out = broadcast(b2) ----------------
__global__ __launch_bounds__(256) void gcn_init(float* __restrict__ h,
                                                const float* __restrict__ b1,
                                                float* __restrict__ out,
                                                const float* __restrict__ b2) {
  long i = (long)blockIdx.x * 256 + threadIdx.x;
  long tot = (long)NN * NHID + (long)NN * NCLASS;
  for (; i < tot; i += (long)gridDim.x * 256) {
    if (i < (long)NN * NHID) {
      h[i] = b1[i & (NHID - 1)];
    } else {
      long j = i - (long)NN * NHID;
      out[j] = b2[j % NCLASS];
    }
  }
}

// ---------------- GEMM1: support = x(f32->bf16) @ W1  (N x 128, f32 out) ----------------
// 256 threads = 8 waves; wave -> 16 rows x 128 cols; K = 512 in 16 steps of 32.
// B panel staged in LDS in WMMA-fragment order:
//   16B chunk j = (((ks*8 + t)*2 + c)*32 + L) at LDS byte j*16
//   source: W1t[(t*16 + L%16)*512 + ks*32 + (L/16)*8 + c*16]
__global__ __launch_bounds__(256) void gcn_gemm1(const float* __restrict__ x,
                                                 const __bf16* __restrict__ W1t,
                                                 float* __restrict__ sup) {
  __shared__ __align__(128) __bf16 Bs[NHID * NFEAT];   // 128 KB
  const int tid  = threadIdx.x;
  const int lane = tid & 31;
  const int wave = tid >> 5;
  const int hi   = lane >> 4;
  const int m0   = blockIdx.x * 128 + wave * 16;
  const int mrow = m0 + (lane & 15);
  const int rrd  = (mrow < NN) ? mrow : (NN - 1);

  // ---- async stage W1t -> LDS (8192 chunks of 16B, 32 per thread) ----
  {
    const uint32_t lbase = (uint32_t)(uintptr_t)(void*)Bs;
#pragma unroll 4
    for (int it = 0; it < 32; ++it) {
      const int j  = it * 256 + tid;
      const int L  = j & 31;
      const int c  = (j >> 5) & 1;
      const int t  = (j >> 6) & 7;
      const int ks = (j >> 9) & 15;
      const long ge = (long)(t * 16 + (L & 15)) * NFEAT + ks * 32 + ((L >> 4) << 3) + c * 16;
      async_g2l_b128(lbase + (uint32_t)j * 16, (const void*)(W1t + ge));
    }
    async_wait_all();
    __syncthreads();
  }

  v8f acc[8];
  const v8f vzero = {0.f, 0.f, 0.f, 0.f, 0.f, 0.f, 0.f, 0.f};
#pragma unroll
  for (int t = 0; t < 8; ++t) acc[t] = vzero;

  const float* xrow = x + (long)rrd * NFEAT;

  for (int ks = 0; ks < 16; ++ks) {
    const int k0 = ks * 32;
    // A fragment (16x32 bf16): lanes 0-15 -> K k0+0..7 / k0+16..23; lanes 16-31 -> +8
    const int c0 = k0 + hi * 8;
    float4 a0 = *(const float4*)(xrow + c0);
    float4 a1 = *(const float4*)(xrow + c0 + 4);
    float4 a2 = *(const float4*)(xrow + c0 + 16);
    float4 a3 = *(const float4*)(xrow + c0 + 20);
    float av[16] = {a0.x, a0.y, a0.z, a0.w, a1.x, a1.y, a1.z, a1.w,
                    a2.x, a2.y, a2.z, a2.w, a3.x, a3.y, a3.z, a3.w};
    v16bf a;
#pragma unroll
    for (int i = 0; i < 16; ++i) a[i] = (__bf16)av[i];

    // preload all 8 B fragments (conflict-free lane-interleaved ds reads)
    BV b[8];
#pragma unroll
    for (int t = 0; t < 8; ++t) {
      b[t].h[0] = *(const v8bf*)(Bs + ((ks * 8 + t) * 2 + 0) * 256 + lane * 8);
      b[t].h[1] = *(const v8bf*)(Bs + ((ks * 8 + t) * 2 + 1) * 256 + lane * 8);
    }
#pragma unroll
    for (int t = 0; t < 8; ++t) {
      acc[t] = __builtin_amdgcn_wmma_f32_16x16x32_bf16(
          false, a, false, b[t].v, (short)0, acc[t], false, false);
    }
  }

  // D layout: VGPR r -> row m0 + r + 8*hi, col = t*16 + lane%16
#pragma unroll
  for (int t = 0; t < 8; ++t) {
    const int n = t * 16 + (lane & 15);
#pragma unroll
    for (int r = 0; r < 8; ++r) {
      int row = m0 + r + hi * 8;
      if (row < NN) sup[(long)row * NHID + n] = acc[t][r];
    }
  }
}

// ---------------- SPMM1: h[row] += w_e * support[col], one edge per wave ----------------
__global__ __launch_bounds__(256) void gcn_spmm1(const long long* __restrict__ idx,
                                                 const float* __restrict__ ew,
                                                 const float* __restrict__ sup,
                                                 float* __restrict__ h) {
  const int lane = threadIdx.x & 31;
  const long e = (long)blockIdx.x * 8 + (threadIdx.x >> 5);
  if (e >= NEDGE) return;
  const int row = (int)idx[e];
  const int col = (int)idx[NEDGE + e];
  const float w = ew[e];
  const float4 v = ((const float4*)(sup + (long)col * NHID))[lane];
  float* dst = h + (long)row * NHID + lane * 4;
  atomicAdd(dst + 0, w * v.x);
  atomicAdd(dst + 1, w * v.y);
  atomicAdd(dst + 2, w * v.z);
  atomicAdd(dst + 3, w * v.w);
}

// ---------------- GEMM2: tmp = relu(h)(bf16) @ W2pad  (N x 48, f32 out) ----------------
// B panel (48x128 = 12KB) staged in LDS, fragment order:
//   chunk j = (((ks*3 + t)*2 + c)*32 + L), ks<4, t<3
__global__ __launch_bounds__(256) void gcn_gemm2(const float* __restrict__ h,
                                                 const __bf16* __restrict__ W2t,
                                                 float* __restrict__ tmp) {
  __shared__ __align__(128) __bf16 Bs2[NCPAD * NHID];  // 12 KB
  const int tid  = threadIdx.x;
  const int lane = tid & 31;
  const int wave = tid >> 5;
  const int hi   = lane >> 4;
  const int m0   = blockIdx.x * 128 + wave * 16;
  const int mrow = m0 + (lane & 15);
  const int rrd  = (mrow < NN) ? mrow : (NN - 1);

  {
    const uint32_t lbase = (uint32_t)(uintptr_t)(void*)Bs2;
#pragma unroll
    for (int it = 0; it < 3; ++it) {
      const int j = it * 256 + tid;   // 768 chunks of 16B
      const int L = j & 31;
      const int p = j >> 5;           // 0..23
      const int c = p & 1;
      const int q = p >> 1;           // 0..11
      const int t = q % 3;
      const int ks = q / 3;
      const long ge = (long)(t * 16 + (L & 15)) * NHID + ks * 32 + ((L >> 4) << 3) + c * 16;
      async_g2l_b128(lbase + (uint32_t)j * 16, (const void*)(W2t + ge));
    }
    async_wait_all();
    __syncthreads();
  }

  v8f acc[3];
  const v8f vzero = {0.f, 0.f, 0.f, 0.f, 0.f, 0.f, 0.f, 0.f};
#pragma unroll
  for (int t = 0; t < 3; ++t) acc[t] = vzero;

  const float* hrow = h + (long)rrd * NHID;

#pragma unroll
  for (int ks = 0; ks < 4; ++ks) {
    const int k0 = ks * 32;
    const int c0 = k0 + hi * 8;
    float4 a0 = *(const float4*)(hrow + c0);
    float4 a1 = *(const float4*)(hrow + c0 + 4);
    float4 a2 = *(const float4*)(hrow + c0 + 16);
    float4 a3 = *(const float4*)(hrow + c0 + 20);
    float av[16] = {a0.x, a0.y, a0.z, a0.w, a1.x, a1.y, a1.z, a1.w,
                    a2.x, a2.y, a2.z, a2.w, a3.x, a3.y, a3.z, a3.w};
    v16bf a;
#pragma unroll
    for (int i = 0; i < 16; ++i) a[i] = (__bf16)fmaxf(av[i], 0.0f);  // fused ReLU

    BV b[3];
#pragma unroll
    for (int t = 0; t < 3; ++t) {
      b[t].h[0] = *(const v8bf*)(Bs2 + ((ks * 3 + t) * 2 + 0) * 256 + lane * 8);
      b[t].h[1] = *(const v8bf*)(Bs2 + ((ks * 3 + t) * 2 + 1) * 256 + lane * 8);
    }
#pragma unroll
    for (int t = 0; t < 3; ++t) {
      acc[t] = __builtin_amdgcn_wmma_f32_16x16x32_bf16(
          false, a, false, b[t].v, (short)0, acc[t], false, false);
    }
  }

#pragma unroll
  for (int t = 0; t < 3; ++t) {
    const int n = t * 16 + (lane & 15);
#pragma unroll
    for (int r = 0; r < 8; ++r) {
      int row = m0 + r + hi * 8;
      if (row < NN) tmp[(long)row * NCPAD + n] = acc[t][r];
    }
  }
}

// ---------------- SPMM2: out[row] += w_e * tmp[col] (40 cols), one edge per wave -------
__global__ __launch_bounds__(256) void gcn_spmm2(const long long* __restrict__ idx,
                                                 const float* __restrict__ ew,
                                                 const float* __restrict__ tmp,
                                                 float* __restrict__ out) {
  const int lane = threadIdx.x & 31;
  const long e = (long)blockIdx.x * 8 + (threadIdx.x >> 5);
  if (e >= NEDGE) return;
  const int row = (int)idx[e];
  const int col = (int)idx[NEDGE + e];
  const float w = ew[e];
  const float* src = tmp + (long)col * NCPAD;
  float* dst = out + (long)row * NCLASS;
  atomicAdd(dst + lane, w * src[lane]);
  if (lane < NCLASS - 32)
    atomicAdd(dst + 32 + lane, w * src[32 + lane]);
}

// ---------------- log_softmax in-place on out, one row per wave ----------------
__global__ __launch_bounds__(256) void gcn_lsm(float* __restrict__ out) {
  const int lane = threadIdx.x & 31;
  const long row = (long)blockIdx.x * 8 + (threadIdx.x >> 5);
  if (row >= NN) return;
  float* p = out + row * NCLASS;
  float v0 = p[lane];
  float v1 = (lane < 8) ? p[32 + lane] : -INFINITY;
  float m = fmaxf(v0, v1);
#pragma unroll
  for (int s = 16; s > 0; s >>= 1) m = fmaxf(m, __shfl_xor(m, s, 32));
  float sum = expf(v0 - m) + ((lane < 8) ? expf(v1 - m) : 0.0f);
#pragma unroll
  for (int s = 16; s > 0; s >>= 1) sum += __shfl_xor(sum, s, 32);
  float lse = m + logf(sum);
  p[lane] = v0 - lse;
  if (lane < 8) p[32 + lane] = v1 - lse;
}

// ---------------- launch ----------------
extern "C" void kernel_launch(void* const* d_in, const int* in_sizes, int n_in,
                              void* d_out, int out_size, void* d_ws, size_t ws_size,
                              hipStream_t stream) {
  const float*     x   = (const float*)d_in[0];
  const long long* idx = (const long long*)d_in[1];
  // d_in[2] (valuess) and d_in[3] (size) unused by forward
  const float* W1 = (const float*)d_in[4];
  const float* b1 = (const float*)d_in[5];
  const float* W2 = (const float*)d_in[6];
  const float* b2 = (const float*)d_in[7];
  const float* ew = (const float*)d_in[8];
  float* out = (float*)d_out;

  // workspace layout (floats):
  //   [0, N*128)           support1  (later reused as tmp N*48)
  //   [N*128, 2*N*128)     h
  //   then bf16: W1t (128*512), W2t (48*128)
  float*  wsf  = (float*)d_ws;
  float*  sup  = wsf;
  float*  h    = wsf + (long)NN * NHID;
  float*  tmp  = wsf;                       // reuse support1 region (dead by then)
  __bf16* W1t  = (__bf16*)(wsf + 2L * NN * NHID);
  __bf16* W2t  = W1t + (long)NHID * NFEAT;

  // 1) weight transpose/convert
  gcn_prep<<<(NHID * NFEAT + 255) / 256, 256, 0, stream>>>(W1, W2, W1t, W2t);
  // 2) init h = b1, out = b2
  gcn_init<<<4096, 256, 0, stream>>>(h, b1, out, b2);
  // 3) support = x @ W1 (bf16 WMMA, async-LDS staged B panel)
  gcn_gemm1<<<(NN + 127) / 128, 256, 0, stream>>>(x, W1t, sup);
  // 4) h += A * support (atomic scatter, L2-resident gathers)
  gcn_spmm1<<<(NEDGE + 7) / 8, 256, 0, stream>>>(idx, ew, sup, h);
  // 5) tmp = relu(h) @ W2 (bf16 WMMA, fused ReLU)
  gcn_gemm2<<<(NN + 127) / 128, 256, 0, stream>>>(h, W2t, tmp);
  // 6) out += A * tmp
  gcn_spmm2<<<(NEDGE + 7) / 8, 256, 0, stream>>>(idx, ew, tmp, out);
  // 7) log_softmax rows in place
  gcn_lsm<<<(NN + 7) / 8, 256, 0, stream>>>(out);
}
